// FrequencySelectivePruner2D_43516608643214
// MI455X (gfx1250) — matrix operand order repaired and checked by reading, not compiled, not run on hardware
//
#include <hip/hip_runtime.h>
#include <hip/hip_bf16.h>
#include <math.h>

// FrequencySelectivePruner2D for MI455X (gfx1250), wave32 + WMMA.
//
// x: [16, 64, 256, 256] f32, tau: scalar f32.
// energy[b,l] = sum_c sum_{m>=1} |DCT2D(64x64) @ patches(64xL)|
// out = x * (energy > tau)
//
// Kernel 1: tensor-core DCT (v_wmma_f32_16x16x32_f16) + energy reduction.
// Kernel 2: bandwidth-bound mask apply.

typedef _Float16 v16h __attribute__((ext_vector_type(16)));
typedef _Float16 v8h  __attribute__((ext_vector_type(8)));
typedef _Float16 v4h  __attribute__((ext_vector_type(4)));
typedef float    v8f  __attribute__((ext_vector_type(8)));
typedef float    v4f  __attribute__((ext_vector_type(4)));

#define IMG_B 16
#define IMG_C 64
#define IMG_H 256
#define IMG_W 256
#define PSZ   8
#define HB    (IMG_H / PSZ)       // 32
#define WBLK  (IMG_W / PSZ)       // 32
#define LPATCH (HB * WBLK)        // 1024 patches per (b,c)
#define LTILE  16                 // patches handled per block
#define NLTILE (LPATCH / LTILE)   // 64 blocks per image

// -------------------------------------------------------------------------
// Kernel 1: per-(b, 16-patch tile) energy via WMMA DCT.
//   block = 256 threads = 8 waves. Wave w handles channels c = 8w..8w+7.
//   D(64x16) = DCT2D(64x64, f16) x P(64x16, f16), fp32 accumulate,
//   energy = sum |D| over m=1..63 and all c.
// -------------------------------------------------------------------------
__global__ __launch_bounds__(256)
void fsp_energy_kernel(const float* __restrict__ x, float* __restrict__ energy)
{
    // DCT2D as f16, row-major [m][n]; A-fragment reads are 16B-contiguous.
    __shared__ __align__(16) _Float16 dct_lds[64 * 64];          // 8 KB
    // Per-wave staging: patch-column-major [j][n] so the B fragment for a
    // lane is 32 contiguous bytes (one v16h load).
    __shared__ __align__(64) _Float16 stage[8][LTILE * 64];      // 16 KB
    __shared__ float wsum[8][LTILE];

    const int tid  = threadIdx.x;
    const int wave = tid >> 5;
    const int lane = tid & 31;

    // ---- build DCT2D = kron(M, M) in LDS (once per block) ----
    // M[k,n] = cos(pi/8*(n+0.5)*k) * (k==0 ? 1/sqrt(8) : 1/2)
    {
        const float PI_8 = 0.39269908169872414f;
        #pragma unroll
        for (int t = 0; t < 16; ++t) {
            int idx = tid * 16 + t;
            int m = idx >> 6, n = idx & 63;
            int k1 = m >> 3, k2 = m & 7, n1 = n >> 3, n2 = n & 7;
            float f1 = __cosf(PI_8 * (n1 + 0.5f) * (float)k1) *
                       (k1 == 0 ? 0.35355339059327373f : 0.5f);
            float f2 = __cosf(PI_8 * (n2 + 0.5f) * (float)k2) *
                       (k2 == 0 ? 0.35355339059327373f : 0.5f);
            dct_lds[idx] = (_Float16)(f1 * f2);
        }
    }
    __syncthreads();

    // ---- preload A fragments (DCT rows), 4 M-tiles x 2 K-steps ----
    // A 16x32 fragment, contiguous k order (same as B): lane m-row = lane%16;
    //   frag halves 0..7  -> K = kstep*32 + (lane/16)*16 + i
    //   frag halves 8..15 -> K = kstep*32 + (lane/16)*16 + 8 + (i-8)
    v16h afrag[4][2];
    {
        const int mrow = lane & 15;
        const int hsel = (lane >> 4) * 16;
        #pragma unroll
        for (int mt = 0; mt < 4; ++mt) {
            #pragma unroll
            for (int ks = 0; ks < 2; ++ks) {
                const _Float16* base = &dct_lds[(mt * 16 + mrow) * 64 + ks * 32 + hsel];
                v8h g0 = *(const v8h*)(base);
                v8h g1 = *(const v8h*)(base + 8);
                v16h a;
                #pragma unroll
                for (int i = 0; i < 8; ++i) { a[i] = g0[i]; a[8 + i] = g1[i]; }
                afrag[mt][ks] = a;
            }
        }
    }

    const int bimg   = blockIdx.x >> 6;        // image
    const int ltile  = blockIdx.x & 63;        // 16-patch tile
    const int hb     = ltile >> 1;             // patch row
    const int wcol0  = (ltile & 1) * 128;      // starting pixel column

    // Staging assignment: lane loads float4 at row offset lane*4 per kh-row.
    // Those 4 floats are all inside patch j = lane/2, kw = (lane&1)*4..+3.
    _Float16* st = stage[wave];
    const int jpatch = lane >> 1;
    const int kwbase = (lane & 1) * 4;

    float eacc = 0.0f;

    for (int ci = 0; ci < 8; ++ci) {
        const int c = wave * 8 + ci;
        const float* rowbase =
            x + (((size_t)(bimg * IMG_C + c) * IMG_H + hb * PSZ) * IMG_W) + wcol0;

        // ---- stage 16 patches (16x64 f32 -> f16, column-major by patch) ----
        #pragma unroll
        for (int kh = 0; kh < 8; ++kh) {
            v4f v = *(const v4f*)(rowbase + kh * IMG_W + lane * 4);
            v4h h;
            h.x = (_Float16)v.x; h.y = (_Float16)v.y;
            h.z = (_Float16)v.z; h.w = (_Float16)v.w;
            *(v4h*)(st + jpatch * 64 + kh * PSZ + kwbase) = h;   // 8B ds_store
        }
        asm volatile("" ::: "memory");   // keep store->load program order

        // ---- B fragments: 16 contiguous halves per lane per K-step ----
        // B 32x16 fragment, contiguous k order (same as A): col N = lane%16; lanes<16 hold K 0..15 of
        // the step, lanes>=16 hold K 16..31 (2 halves per VGPR, in order).
        const _Float16* bbase = st + (lane & 15) * 64 + (lane >> 4) * 16;
        v16h b0 = *(const v16h*)(bbase);        // K in [ 0,32)
        v16h b1 = *(const v16h*)(bbase + 32);   // K in [32,64)

        // ---- 4 M-tiles x 2 K-steps of WMMA, |.|-reduce immediately ----
        #pragma unroll
        for (int mt = 0; mt < 4; ++mt) {
            v8f d = {};
            d = __builtin_amdgcn_wmma_f32_16x16x32_f16(
                    false, afrag[mt][0], false, b0, (short)0, d, false, false);
            d = __builtin_amdgcn_wmma_f32_16x16x32_f16(
                    false, afrag[mt][1], false, b1, (short)0, d, false, false);
            // C/D layout: lanes<16: VGPR r -> m = mt*16 + r (col = lane);
            //             lanes>=16: m = mt*16 + r + 8 (col = lane-16).
            float s = 0.0f;
            #pragma unroll
            for (int r = 0; r < 8; ++r) s += fabsf(d[r]);
            if (mt == 0 && lane < 16) s -= fabsf(d[0]);   // drop DC (m==0)
            eacc += s;
        }
    }

    // Fold the two lane halves (same column j = lane%16, disjoint m sets).
    eacc += __shfl_xor(eacc, 16, 32);
    if (lane < 16) wsum[wave][lane] = eacc;
    __syncthreads();

    // Fold the 8 waves (channels) and emit 16 energies for this tile.
    if (tid < 16) {
        float t = 0.0f;
        #pragma unroll
        for (int w = 0; w < 8; ++w) t += wsum[w][tid];
        energy[bimg * LPATCH + ltile * LTILE + tid] = t;
    }
}

// -------------------------------------------------------------------------
// Kernel 2: out = x * (energy[b, patch(h,w)] > tau). Pure bandwidth.
// Each float4 (16B aligned) lies inside a single 8-wide patch.
// -------------------------------------------------------------------------
__global__ __launch_bounds__(256)
void fsp_mask_kernel(const float* __restrict__ x,
                     const float* __restrict__ energy,
                     const float* __restrict__ tau_p,
                     float* __restrict__ out, int n4)
{
    int idx = blockIdx.x * 256 + threadIdx.x;
    if (idx >= n4) return;
    const float tau = tau_p[0];

    size_t flat = (size_t)idx * 4;
    int w  = (int)(flat & (IMG_W - 1));
    int h  = (int)((flat >> 8) & (IMG_H - 1));
    int bc = (int)(flat >> 16);
    int b  = bc >> 6;
    int l  = (h >> 3) * WBLK + (w >> 3);

    float e = energy[b * LPATCH + l];
    float m = (e > tau) ? 1.0f : 0.0f;

    v4f v = *(const v4f*)(x + flat);
    v.x *= m; v.y *= m; v.z *= m; v.w *= m;
    *(v4f*)(out + flat) = v;
}

// -------------------------------------------------------------------------
extern "C" void kernel_launch(void* const* d_in, const int* in_sizes, int n_in,
                              void* d_out, int out_size, void* d_ws, size_t ws_size,
                              hipStream_t stream)
{
    (void)in_sizes; (void)n_in; (void)out_size; (void)ws_size;
    const float* x   = (const float*)d_in[0];
    const float* tau = (const float*)d_in[1];
    float* out    = (float*)d_out;
    float* energy = (float*)d_ws;            // 16*1024 floats = 64 KB scratch

    // Kernel 1 fully writes energy each call (no init pass, no atomics).
    fsp_energy_kernel<<<IMG_B * NLTILE, 256, 0, stream>>>(x, energy);

    const int n4 = (IMG_B * IMG_C * IMG_H * IMG_W) / 4;   // 16,777,216
    fsp_mask_kernel<<<n4 / 256, 256, 0, stream>>>(x, energy, tau, out, n4);
}
